// seqGRU_48292612276419
// MI455X (gfx1250) — matrix operand order repaired and checked by reading, not compiled, not run on hardware
//
#include <hip/hip_runtime.h>
#include <math.h>

// ---------------------------------------------------------------------------
// seqGRU on MI455X (gfx1250): bidirectional 2-layer GRU, H=128, B=128, T=2048
//   K0  pack w_hh (l0f,l0b,l1f) and w_ih_l1f into WMMA B-fragment layout (f16)
//   K1  layer-0 scan, both dirs: w_hh resident in VGPRs, per-step 16x384x128
//       GEMM via v_wmma_f32_16x16x32_f16; hidden state kept in LDS in a
//       fragment-native swizzle so A-frags are single contiguous loads.
//   K2  gx1f = y0 @ w_ih_l1f^T + b : WMMA GEMM (262144 x 384 x 256)
//   K3  layer-1 forward scan (gx streamed + global_prefetch)
//   K4  layer-1 backward needs ONE step (h0=0 => gh=b_hh); fused with FC
//
// Swizzled f16 layouts (fragment-native, per 16-row tile):
//   A-frag for WMMA_16x16x32: lane(half,r16) needs K = kb*32 + part*16 +
//   half*8 + e  (part=0 -> elems 0..7, part=1 -> elems 8..15).
//   We store value[row][k] at chunk ((row*KB + kb)*2 + half)*16 + part*8 + e,
//   so each lane's fragment is one contiguous 32B chunk and the gate phase's
//   aligned 8-value groups are single 16B stores.
// ---------------------------------------------------------------------------

typedef _Float16 v16h __attribute__((ext_vector_type(16)));
typedef _Float16 h8   __attribute__((ext_vector_type(8)));
typedef float    v8f  __attribute__((ext_vector_type(8)));

#define WMMA16(A, Bm, C) __builtin_amdgcn_wmma_f32_16x16x32_f16( \
    false, (A), false, (Bm), (short)0, (C), false, false)

__device__ __forceinline__ float sigm(float x) { return 1.0f / (1.0f + __expf(-x)); }

#define GRU_B 128
#define GRU_T 2048
#define GRU_H 128
#define GRU_G 384          // 3*H
#define GRU_DIN 5
#define GRU_DL1 256

// --- K0a: pack w_hh (384 x 128) f32 -> [tile24][kb4][lane32][e16] f16 -------
__global__ void pack_whh(const float* __restrict__ w, _Float16* __restrict__ out) {
  int idx = blockIdx.x * 256 + threadIdx.x;       // 24*4*32 = 3072
  if (idx >= 24 * 4 * 32) return;
  int lane = idx & 31, kb = (idx >> 5) & 3, tile = idx >> 7;
  int col = tile * 16 + (lane & 15);
  int k0  = kb * 32 + (lane >> 4) * 8;
  _Float16* o = out + (size_t)idx * 16;
  #pragma unroll
  for (int e = 0; e < 16; ++e) o[e] = (_Float16)w[col * GRU_H + k0 + (e & 7) + (e >> 3) * 16];
}

// --- K0b: pack w_ih_l1f (384 x 256) -> [tile24][kb8][lane32][e16] -----------
__global__ void pack_wih1(const float* __restrict__ w, _Float16* __restrict__ out) {
  int idx = blockIdx.x * 256 + threadIdx.x;       // 24*8*32 = 6144
  if (idx >= 24 * 8 * 32) return;
  int lane = idx & 31, kb = (idx >> 5) & 7, tile = idx >> 8;
  int col = tile * 16 + (lane & 15);
  int k0  = kb * 32 + (lane >> 4) * 8;
  _Float16* o = out + (size_t)idx * 16;
  #pragma unroll
  for (int e = 0; e < 16; ++e) o[e] = (_Float16)w[col * GRU_DL1 + k0 + (e & 7) + (e >> 3) * 16];
}

// --- K1: layer-0 scan, grid (8 batch-tiles, 2 dirs), 256 thr (8 waves) ------
__global__ __launch_bounds__(256)
void scan_l0(const float* __restrict__ seq,
             const float* __restrict__ wih_f, const float* __restrict__ bih_f,
             const float* __restrict__ bhh_f,
             const float* __restrict__ wih_b, const float* __restrict__ bih_b,
             const float* __restrict__ bhh_b,
             const _Float16* __restrict__ whh_pack,   // 2 dirs, stride 49152
             _Float16* __restrict__ y0s) {            // swizzled y0, [m][kb8][half2][16]
  const int dir  = blockIdx.y;
  const int b0   = blockIdx.x * 16;
  const int tid  = threadIdx.x;
  const int lane = tid & 31, wave = tid >> 5;
  const int half = lane >> 4, r16 = lane & 15;

  __shared__ __align__(16) float    gxs[16][GRU_G];
  __shared__ __align__(16) float    ghs[16][GRU_G];
  __shared__ __align__(16) float    hf [16][GRU_H];
  __shared__ __align__(16) _Float16 hswz[16 * 4 * 2 * 16];   // fragment-native h
  __shared__ float wih_s[GRU_G * GRU_DIN];
  __shared__ float bih_s[GRU_G], bhh_s[GRU_G];

  const float* wih = dir ? wih_b : wih_f;
  const float* bih = dir ? bih_b : bih_f;
  const float* bhh = dir ? bhh_b : bhh_f;

  for (int i = tid; i < GRU_G * GRU_DIN; i += 256) wih_s[i] = wih[i];
  for (int i = tid; i < GRU_G; i += 256) { bih_s[i] = bih[i]; bhh_s[i] = bhh[i]; }
  for (int i = tid; i < 16 * GRU_H; i += 256) {
    (&hf[0][0])[i] = 0.0f; hswz[i] = (_Float16)0.0f;
  }

  // w_hh B-fragments: resident in VGPRs for the whole scan (3 col-tiles/wave)
  const _Float16* wp = whh_pack + (size_t)dir * (24 * 4 * 32 * 16);
  v16h Bf[3][4];
  #pragma unroll
  for (int i = 0; i < 3; ++i) {
    const int tile = wave + i * 8;
    #pragma unroll
    for (int kb = 0; kb < 4; ++kb)
      Bf[i][kb] = *(const v16h*)(wp + ((size_t)(tile * 4 + kb) * 32 + lane) * 16);
  }
  __syncthreads();

  const int grow  = tid >> 4;             // row (0..15) for gx + gates
  const int q     = tid & 15;             // j0 = 8*q
  const int gcol0 = q * 24;               // gx column chunk
  const int j0    = q * 8;                // gate hidden-unit chunk
  // swizzle coords for this thread's 8-value h group
  const int g_kb = q >> 2, g_half = q & 1, g_part = (q >> 1) & 1;
  const int h_off = ((grow * 4 + g_kb) * 2 + g_half) * 16 + g_part * 8;
  // y0 swizzle: column c = dir*128 + j0; q2 = c/8 = dir*16 + q (kb over 8)
  const int q2 = dir * 16 + q;
  const int y_kb = q2 >> 2, y_half = q2 & 1, y_part = (q2 >> 1) & 1;

  for (int ts = 0; ts < GRU_T; ++ts) {
    const int t = dir ? (GRU_T - 1 - ts) : ts;

    // ---- gh = h @ w_hh^T : 12 WMMAs/wave (A-frag = one contiguous chunk) ----
    v16h Af[4];
    #pragma unroll
    for (int kb = 0; kb < 4; ++kb)
      Af[kb] = *(const v16h*)&hswz[((r16 * 4 + kb) * 2 + half) * 16];
    v8f acc[3] = {};
    #pragma unroll
    for (int i = 0; i < 3; ++i)
      #pragma unroll
      for (int kb = 0; kb < 4; ++kb)
        acc[i] = WMMA16(Af[kb], Bf[i][kb], acc[i]);
    #pragma unroll
    for (int i = 0; i < 3; ++i) {
      const int col = (wave + i * 8) * 16 + r16;
      #pragma unroll
      for (int e = 0; e < 8; ++e) ghs[e + 8 * half][col] = acc[i][e];
    }

    // ---- gx = x @ w_ih^T + b_ih  (K=5, VALU) ------------------------------
    {
      const float* sp = seq + ((size_t)(b0 + grow) * GRU_T + t) * GRU_DIN;
      const float s0 = sp[0], s1 = sp[1], s2 = sp[2], s3 = sp[3], s4 = sp[4];
      #pragma unroll
      for (int gg = 0; gg < 24; ++gg) {
        const int g = gcol0 + gg;
        const float* w5 = &wih_s[g * GRU_DIN];
        gxs[grow][g] = bih_s[g] + s0 * w5[0] + s1 * w5[1] + s2 * w5[2]
                                + s3 * w5[3] + s4 * w5[4];
      }
    }
    __syncthreads();

    // ---- gates + hidden update -------------------------------------------
    h8 hv;
    #pragma unroll
    for (int jj = 0; jj < 8; ++jj) {
      const int j = j0 + jj;
      const float gr = gxs[grow][j]           + ghs[grow][j]           + bhh_s[j];
      const float gz = gxs[grow][GRU_H + j]   + ghs[grow][GRU_H + j]   + bhh_s[GRU_H + j];
      const float gn = gxs[grow][2*GRU_H + j];
      const float hn_ = ghs[grow][2*GRU_H + j] + bhh_s[2*GRU_H + j];
      const float r = sigm(gr), z = sigm(gz);
      const float n = tanhf(gn + r * hn_);
      const float hnew = (1.0f - z) * n + z * hf[grow][j];
      hf[grow][j] = hnew;
      hv[jj] = (_Float16)hnew;
    }
    *(h8*)&hswz[h_off] = hv;                                // one 16B LDS store
    const size_t m = (size_t)(b0 + grow) * GRU_T + t;
    *(h8*)&y0s[((m * 8 + y_kb) * 2 + y_half) * 16 + y_part * 8] = hv;  // 16B global
    __syncthreads();
  }
}

// --- K2: gx1f = y0 @ w_ih_l1f^T + b_ih  (M=262144, N=384, K=256) ------------
__global__ __launch_bounds__(256)
void gemm_gx1(const _Float16* __restrict__ y0s, const _Float16* __restrict__ wpack,
              const float* __restrict__ bih, float* __restrict__ gx1) {
  const int tid  = threadIdx.x;
  const int lane = tid & 31, wave = tid >> 5;
  const int half = lane >> 4, r16 = lane & 15;
  const int m0   = blockIdx.x * 128 + wave * 16;
  const int nt   = blockIdx.y;

  v16h Bf[8];
  #pragma unroll
  for (int kb = 0; kb < 8; ++kb)
    Bf[kb] = *(const v16h*)(wpack + ((size_t)(nt * 8 + kb) * 32 + lane) * 16);

  v8f acc = {};
  #pragma unroll
  for (int kb = 0; kb < 8; ++kb) {
    // swizzled y0: lane's A fragment is one contiguous 32B chunk
    const v16h Af = *(const v16h*)&y0s[(((size_t)(m0 + r16) * 8 + kb) * 2 + half) * 16];
    acc = WMMA16(Af, Bf[kb], acc);
  }
  const int col  = nt * 16 + r16;
  const float bb = bih[col];
  #pragma unroll
  for (int e = 0; e < 8; ++e)
    gx1[(size_t)(m0 + e + 8 * half) * GRU_G + col] = acc[e] + bb;
}

// --- K3: layer-1 forward scan; keeps only final hidden state ----------------
__global__ __launch_bounds__(256)
void scan_l1(const float* __restrict__ gx1, const _Float16* __restrict__ whh_pack,
             const float* __restrict__ bhh, float* __restrict__ hlast) {
  const int b0   = blockIdx.x * 16;
  const int tid  = threadIdx.x;
  const int lane = tid & 31, wave = tid >> 5;
  const int half = lane >> 4, r16 = lane & 15;

  __shared__ __align__(16) float    ghs[16][GRU_G];
  __shared__ __align__(16) float    hf [16][GRU_H];
  __shared__ __align__(16) _Float16 hswz[16 * 4 * 2 * 16];
  __shared__ float bhh_s[GRU_G];

  for (int i = tid; i < GRU_G; i += 256) bhh_s[i] = bhh[i];
  for (int i = tid; i < 16 * GRU_H; i += 256) {
    (&hf[0][0])[i] = 0.0f; hswz[i] = (_Float16)0.0f;
  }

  v16h Bf[3][4];
  #pragma unroll
  for (int i = 0; i < 3; ++i) {
    const int tile = wave + i * 8;
    #pragma unroll
    for (int kb = 0; kb < 4; ++kb)
      Bf[i][kb] = *(const v16h*)(whh_pack + ((size_t)(tile * 4 + kb) * 32 + lane) * 16);
  }
  __syncthreads();

  const int grow = tid >> 4;
  const int q    = tid & 15;
  const int j0   = q * 8;
  const int g_kb = q >> 2, g_half = q & 1, g_part = (q >> 1) & 1;
  const int h_off = ((grow * 4 + g_kb) * 2 + g_half) * 16 + g_part * 8;

  for (int t = 0; t < GRU_T; ++t) {
    v16h Af[4];
    #pragma unroll
    for (int kb = 0; kb < 4; ++kb)
      Af[kb] = *(const v16h*)&hswz[((r16 * 4 + kb) * 2 + half) * 16];
    v8f acc[3] = {};
    #pragma unroll
    for (int i = 0; i < 3; ++i)
      #pragma unroll
      for (int kb = 0; kb < 4; ++kb)
        acc[i] = WMMA16(Af[kb], Bf[i][kb], acc[i]);
    #pragma unroll
    for (int i = 0; i < 3; ++i) {
      const int col = (wave + i * 8) * 16 + r16;
      #pragma unroll
      for (int e = 0; e < 8; ++e) ghs[e + 8 * half][col] = acc[i][e];
    }
    __syncthreads();

    const float* gp = gx1 + ((size_t)(b0 + grow) * GRU_T + t) * GRU_G;
    if (t + 1 < GRU_T) __builtin_prefetch(gp + GRU_G, 0, 1);  // global_prefetch_b8
    h8 hv;
    #pragma unroll
    for (int jj = 0; jj < 8; ++jj) {
      const int j = j0 + jj;
      const float gr = gp[j]           + ghs[grow][j]           + bhh_s[j];
      const float gz = gp[GRU_H + j]   + ghs[grow][GRU_H + j]   + bhh_s[GRU_H + j];
      const float gn = gp[2*GRU_H + j];
      const float hn_ = ghs[grow][2*GRU_H + j] + bhh_s[2*GRU_H + j];
      const float r = sigm(gr), z = sigm(gz);
      const float n = tanhf(gn + r * hn_);
      const float hnew = (1.0f - z) * n + z * hf[grow][j];
      hf[grow][j] = hnew;
      hv[jj] = (_Float16)hnew;
    }
    *(h8*)&hswz[h_off] = hv;
    __syncthreads();
  }
  #pragma unroll
  for (int jj = 0; jj < 8; ++jj)
    hlast[(size_t)(b0 + grow) * GRU_H + j0 + jj] = hf[grow][j0 + jj];
}

// --- K4: layer-1 backward single step (h0=0 => gh=b_hh) + FC ---------------
__global__ __launch_bounds__(256)
void final_fc(const _Float16* __restrict__ y0s,
              const float* __restrict__ wih1b, const float* __restrict__ bih1b,
              const float* __restrict__ bhh1b, const float* __restrict__ hlast,
              const float* __restrict__ fcw, const float* __restrict__ fcb,
              float* __restrict__ out) {
  const int b = blockIdx.x, tid = threadIdx.x;
  __shared__ float y0r[GRU_DL1];
  __shared__ float gxb[GRU_G];
  __shared__ float hb[GRU_H];
  __shared__ float red[256];

  // un-swizzle one y0 row (b, t = T-1)
  {
    const size_t m = (size_t)b * GRU_T + (GRU_T - 1);
    const int c = tid;
    const int kb = c >> 5, half = (c >> 3) & 1, part = (c >> 4) & 1, e = c & 7;
    y0r[c] = (float)y0s[((m * 8 + kb) * 2 + half) * 16 + part * 8 + e];
  }
  __syncthreads();

  for (int g = tid; g < GRU_G; g += 256) {
    float s = bih1b[g];
    const float* w = wih1b + (size_t)g * GRU_DL1;
    for (int k = 0; k < GRU_DL1; ++k) s += y0r[k] * w[k];
    gxb[g] = s;
  }
  __syncthreads();

  if (tid < GRU_H) {
    const float r = sigm(gxb[tid] + bhh1b[tid]);
    const float z = sigm(gxb[GRU_H + tid] + bhh1b[GRU_H + tid]);
    const float n = tanhf(gxb[2*GRU_H + tid] + r * bhh1b[2*GRU_H + tid]);
    hb[tid] = (1.0f - z) * n;        // + z*h0, h0 == 0
  }
  __syncthreads();

  const float v = (tid < GRU_H) ? hlast[(size_t)b * GRU_H + tid] : hb[tid - GRU_H];
  red[tid] = fcw[tid] * v;
  __syncthreads();
  for (int s = 128; s > 0; s >>= 1) {
    if (tid < s) red[tid] += red[tid + s];
    __syncthreads();
  }
  if (tid == 0) out[b] = red[0] + fcb[0];
}

// ---------------------------------------------------------------------------
extern "C" void kernel_launch(void* const* d_in, const int* in_sizes, int n_in,
                              void* d_out, int out_size, void* d_ws, size_t ws_size,
                              hipStream_t stream) {
  const float* seq      = (const float*)d_in[0];
  const float* w_ih_l0f = (const float*)d_in[1];
  const float* w_hh_l0f = (const float*)d_in[2];
  const float* b_ih_l0f = (const float*)d_in[3];
  const float* b_hh_l0f = (const float*)d_in[4];
  const float* w_ih_l0b = (const float*)d_in[5];
  const float* w_hh_l0b = (const float*)d_in[6];
  const float* b_ih_l0b = (const float*)d_in[7];
  const float* b_hh_l0b = (const float*)d_in[8];
  const float* w_ih_l1f = (const float*)d_in[9];
  const float* w_hh_l1f = (const float*)d_in[10];
  const float* b_ih_l1f = (const float*)d_in[11];
  const float* b_hh_l1f = (const float*)d_in[12];
  const float* w_ih_l1b = (const float*)d_in[13];
  const float* w_hh_l1b = (const float*)d_in[14];  // unused: l1b is 1 step, gh=b_hh
  const float* b_ih_l1b = (const float*)d_in[15];
  const float* b_hh_l1b = (const float*)d_in[16];
  const float* fc_w     = (const float*)d_in[17];
  const float* fc_b     = (const float*)d_in[18];
  (void)w_hh_l1b; (void)in_sizes; (void)n_in; (void)out_size; (void)ws_size;

  // workspace layout (byte offsets, all 256-aligned)
  char* ws = (char*)d_ws;
  const size_t WHH_STRIDE = (size_t)24 * 4 * 32 * 16;             // halfs per dir
  _Float16* whh_pack = (_Float16*)(ws + 0);                        // 3 dirs
  _Float16* wih1pack = (_Float16*)(ws + 294912);
  _Float16* y0s      = (_Float16*)(ws + 491520);                   // B*T*256 f16 (swizzled)
  float*    gx1      = (float*)   (ws + 134709248);                // B*T*384 f32
  float*    hlast    = (float*)   (ws + 537362432);                // 128*128 f32
  float*    out      = (float*)d_out;

  pack_whh<<<12, 256, 0, stream>>>(w_hh_l0f, whh_pack);
  pack_whh<<<12, 256, 0, stream>>>(w_hh_l0b, whh_pack + WHH_STRIDE);
  pack_whh<<<12, 256, 0, stream>>>(w_hh_l1f, whh_pack + 2 * WHH_STRIDE);
  pack_wih1<<<24, 256, 0, stream>>>(w_ih_l1f, wih1pack);

  scan_l0<<<dim3(8, 2), 256, 0, stream>>>(seq,
      w_ih_l0f, b_ih_l0f, b_hh_l0f,
      w_ih_l0b, b_ih_l0b, b_hh_l0b,
      whh_pack, y0s);

  gemm_gx1<<<dim3((GRU_B * GRU_T) / 128, 24), 256, 0, stream>>>(y0s, wih1pack,
                                                                b_ih_l1f, gx1);

  scan_l1<<<8, 256, 0, stream>>>(gx1, whh_pack + 2 * WHH_STRIDE, b_hh_l1f, hlast);

  final_fc<<<128, 256, 0, stream>>>(y0s, w_ih_l1b, b_ih_l1b, b_hh_l1b,
                                    hlast, fc_w, fc_b, out);
}
